// MultiHeadedAttention_5394478923927
// MI455X (gfx1250) — matrix operand order repaired and checked by reading, not compiled, not run on hardware
//
#include <hip/hip_runtime.h>

#define HEADS 16
#define DK 64
#define LDT 40  // LDS row stride in halfs (multiple of 8 -> 16B aligned rows)

typedef __attribute__((ext_vector_type(16))) _Float16 v16h;
typedef __attribute__((ext_vector_type(8)))  _Float16 v8h;
typedef __attribute__((ext_vector_type(8)))  float    v8f;

// ---------------------------------------------------------------------------
// helpers
// ---------------------------------------------------------------------------
__device__ __forceinline__ v8f vzero() {
  v8f z;
#pragma unroll
  for (int i = 0; i < 8; ++i) z[i] = 0.0f;
  return z;
}

__device__ __forceinline__ v16h frag_cat(v8h lo, v8h hi) {
  v16h r;
#pragma unroll
  for (int i = 0; i < 8; ++i) { r[i] = lo[i]; r[i + 8] = hi[i]; }
  return r;
}

// A-matrix fragment (16 rows x 32 K, f16). ISA layout: lane L(0..15) holds row
// M=L, K in {kg*8+0..7, 16+kg*8+0..7} with kg = lane>>4. base -> A[0][0],
// row stride ld (halfs). Two 16B loads per lane.
__device__ __forceinline__ v16h load_frag_a(const _Float16* base, int ld) {
  const int lane = threadIdx.x & 31;
  const _Float16* p = base + (lane & 15) * ld + ((lane >> 4) << 3);
  return frag_cat(*(const v8h*)p, *(const v8h*)(p + 16));
}

// B-matrix fragment (32 K x 16 cols, f16) from TRANSPOSED storage:
// element B[k][n] = base[n*ld + k]. Lane L holds col N=L&15, K in {kg*8+0..7, 16+kg*8+0..7}, kg = L>>4.
__device__ __forceinline__ v16h load_frag_b(const _Float16* base, int ld) {
  const int lane = threadIdx.x & 31;
  const _Float16* p = base + (lane & 15) * ld + ((lane >> 4) << 3);
  return frag_cat(*(const v8h*)p, *(const v8h*)(p + 16));
}

__device__ __forceinline__ v8f wmma16(v16h a, v16h b, v8f c) {
  // (neg_a, A, neg_b, B, c_mod, C, reuse_a, reuse_b)
  return __builtin_amdgcn_wmma_f32_16x16x32_f16(false, a, false, b, (short)0, c,
                                                false, false);
}

// ds_swizzle XOR butterfly inside each 16-lane half (matches C-layout rows)
#define SWZ_XOR(x, m)                                                          \
  __builtin_bit_cast(float, __builtin_amdgcn_ds_swizzle(                       \
                                __builtin_bit_cast(int, (x)), (0x1f | ((m) << 10))))

__device__ __forceinline__ float rowmax16(float t) {
  t = fmaxf(t, SWZ_XOR(t, 1));
  t = fmaxf(t, SWZ_XOR(t, 2));
  t = fmaxf(t, SWZ_XOR(t, 4));
  t = fmaxf(t, SWZ_XOR(t, 8));
  return t;
}
__device__ __forceinline__ float rowsum16(float t) {
  t += SWZ_XOR(t, 1);
  t += SWZ_XOR(t, 2);
  t += SWZ_XOR(t, 4);
  t += SWZ_XOR(t, 8);
  return t;
}

// ---------------------------------------------------------------------------
// Kernel 1: P = X @ W^T + b, f32 in, f16 out packed [B, H, S, DK]
// block = 256 threads (8 waves), tile 128x128, K-step 32.
// Wave (mw, nw) with mw = wave>>1, nw = wave&1 computes a 32x64 sub-tile:
// 2 A-fragments x 4 shared B-fragments -> 8 back-to-back WMMAs per interval.
// ---------------------------------------------------------------------------
__global__ __launch_bounds__(256) void qkv_proj_kernel(
    const float* __restrict__ X, const float* __restrict__ W,
    const float* __restrict__ bias, _Float16* __restrict__ out,
    int M, int N, int K, int S) {
  __shared__ __align__(16) _Float16 As[128 * LDT];
  __shared__ __align__(16) _Float16 Ws[128 * LDT];

  const int ntile = blockIdx.x * 128;
  const int mtile = blockIdx.y * 128;
  const int t = threadIdx.x;
  const int wave = t >> 5;
  const int lane = t & 31;
  const int mw = wave >> 1;  // 0..3 : 32-row group
  const int nw = wave & 1;   // 0..1 : 64-col group

  v8f acc[2][4];
#pragma unroll
  for (int g = 0; g < 2; ++g)
#pragma unroll
    for (int i = 0; i < 4; ++i) acc[g][i] = vzero();

  const int srow = t >> 1;        // 0..127
  const int scol = (t & 1) << 4;  // 0 or 16

  for (int kb = 0; kb < K; kb += 32) {
    // stage + convert A and W 128x32 tiles to f16 LDS
    {
      const float* sa = X + (size_t)(mtile + srow) * K + kb + scol;
      const float* sw = W + (size_t)(ntile + srow) * K + kb + scol;
      _Float16* da = As + srow * LDT + scol;
      _Float16* dw = Ws + srow * LDT + scol;
#pragma unroll
      for (int i = 0; i < 16; i += 4) {
        float4 fa = *(const float4*)(sa + i);
        float4 fw = *(const float4*)(sw + i);
        da[i + 0] = (_Float16)fa.x; da[i + 1] = (_Float16)fa.y;
        da[i + 2] = (_Float16)fa.z; da[i + 3] = (_Float16)fa.w;
        dw[i + 0] = (_Float16)fw.x; dw[i + 1] = (_Float16)fw.y;
        dw[i + 2] = (_Float16)fw.z; dw[i + 3] = (_Float16)fw.w;
      }
      if (kb + 32 < K) {
        __builtin_prefetch(sa + 32, 0, 0);  // global_prefetch_b8
        __builtin_prefetch(sw + 32, 0, 0);
      }
    }
    __syncthreads();

    v16h a0 = load_frag_a(As + (mw * 32) * LDT, LDT);
    v16h a1 = load_frag_a(As + (mw * 32 + 16) * LDT, LDT);
#pragma unroll
    for (int ns = 0; ns < 4; ++ns) {
      v16h b = load_frag_b(Ws + (nw * 64 + ns * 16) * LDT, LDT);  // B = W^T
      acc[0][ns] = wmma16(a0, b, acc[0][ns]);
      acc[1][ns] = wmma16(a1, b, acc[1][ns]);
    }
    __syncthreads();
  }

  // epilogue: bias, convert, scatter to head-major [B, H, S, DK]
  const int hi = lane >> 4;
  const int col = lane & 15;
#pragma unroll
  for (int g = 0; g < 2; ++g) {
    const int mrow = mtile + mw * 32 + g * 16;
#pragma unroll
    for (int ns = 0; ns < 4; ++ns) {
      const int n = ntile + nw * 64 + ns * 16 + col;
      const float bv = bias[n];
      const int h = n >> 6, d = n & 63;
#pragma unroll
      for (int r = 0; r < 8; ++r) {
        const int m = mrow + (hi << 3) + r;
        const int b = m / S, s = m - b * S;
        out[(((size_t)b * HEADS + h) * S + s) * DK + d] =
            (_Float16)(acc[g][ns][r] + bv);
      }
    }
  }
}

// ---------------------------------------------------------------------------
// Kernel 2: flash attention, f16 in/out, layouts [B, H, S, DK]
// block = 256 threads (8 waves); block handles 128 queries of one (b,h);
// each wave owns 16 queries; key blocks of 32; online softmax.
// ---------------------------------------------------------------------------
__global__ __launch_bounds__(256) void flash_attn_kernel(
    const _Float16* __restrict__ Q, const _Float16* __restrict__ Kh,
    const _Float16* __restrict__ V, _Float16* __restrict__ O, int S) {
  __shared__ __align__(16) _Float16 Vt[DK * LDT];      // transposed V tile [d][key]
  __shared__ __align__(16) _Float16 Pb[8 * 16 * LDT];  // per-wave P tiles

  const size_t base = (size_t)blockIdx.y * S * DK;
  const _Float16* q = Q + base;
  const _Float16* k = Kh + base;
  const _Float16* v = V + base;
  _Float16* o = O + base;

  const int t = threadIdx.x;
  const int wave = t >> 5;
  const int lane = t & 31;
  const int hi = lane >> 4;
  const int col = lane & 15;
  const int qbase = blockIdx.x * 128 + wave * 16;

  // Q fragments: 16 rows x d_k=64 split into two 16x32 A fragments
  const v16h qa0 = load_frag_a(q + (size_t)qbase * DK, DK);
  const v16h qa1 = load_frag_a(q + (size_t)qbase * DK + 32, DK);

  v8f acc[4];
#pragma unroll
  for (int i = 0; i < 4; ++i) acc[i] = vzero();
  float mrow[8], lrow[8];
#pragma unroll
  for (int r = 0; r < 8; ++r) { mrow[r] = -1e30f; lrow[r] = 0.0f; }

  _Float16* Pw = Pb + wave * 16 * LDT;
  const int vkey = t >> 3;       // 0..31 : key within block
  const int vdb = (t & 7) << 3;  // 0..56 : dim chunk

  for (int kb = 0; kb < S; kb += 32) {
    __syncthreads();  // previous iteration's Vt consumers done
    // stage transposed V tile: Vt[d][key] = V[kb+key][d]
    {
      v8h val = *(const v8h*)(v + (size_t)(kb + vkey) * DK + vdb);
#pragma unroll
      for (int i = 0; i < 8; ++i) Vt[(vdb + i) * LDT + vkey] = val[i];
    }
    if (kb + 32 < S)
      __builtin_prefetch(v + (size_t)(kb + 32 + vkey) * DK + vdb, 0, 0);
    __syncthreads();

    // scores: 16 queries x 32 keys (K^T B-fragments straight from global:
    // d_k is the contiguous axis of [S, DK], which is exactly B^T storage)
    v8f c0 = vzero(), c1 = vzero();
    {
      const _Float16* kblk = k + (size_t)kb * DK;
      c0 = wmma16(qa0, load_frag_b(kblk, DK), c0);
      c0 = wmma16(qa1, load_frag_b(kblk + 32, DK), c0);
      c1 = wmma16(qa0, load_frag_b(kblk + 16 * DK, DK), c1);
      c1 = wmma16(qa1, load_frag_b(kblk + 16 * DK + 32, DK), c1);
    }

    // online softmax (row = r + 8*hi lives across the 16 lanes of a half)
#pragma unroll
    for (int r = 0; r < 8; ++r) {
      const float s0 = c0[r] * 0.125f;  // 1/sqrt(64)
      const float s1 = c1[r] * 0.125f;
      const float mx = rowmax16(fmaxf(s0, s1));
      const float mnew = fmaxf(mrow[r], mx);
      const float scale = __expf(mrow[r] - mnew);
      mrow[r] = mnew;
      const float p0 = __expf(s0 - mnew);
      const float p1 = __expf(s1 - mnew);
      lrow[r] = lrow[r] * scale + rowsum16(p0 + p1);
#pragma unroll
      for (int n2 = 0; n2 < 4; ++n2) acc[n2][r] = acc[n2][r] * scale;
      _Float16* pr = Pw + ((hi << 3) + r) * LDT;  // C-layout -> row-major P
      pr[col] = (_Float16)p0;
      pr[col + 16] = (_Float16)p1;
    }

    // ctx += P(16x32) x V(32x64): A from LDS P, B from transposed Vt
    v16h pa = load_frag_a(Pw, LDT);
#pragma unroll
    for (int n2 = 0; n2 < 4; ++n2) {
      v16h vb = load_frag_b(Vt + n2 * 16 * LDT, LDT);
      acc[n2] = wmma16(pa, vb, acc[n2]);
    }
  }

  // normalize + store ctx (f16, head-major)
  float inv[8];
#pragma unroll
  for (int r = 0; r < 8; ++r) inv[r] = 1.0f / lrow[r];
#pragma unroll
  for (int n2 = 0; n2 < 4; ++n2) {
#pragma unroll
    for (int r = 0; r < 8; ++r) {
      const int row = qbase + (hi << 3) + r;
      o[(size_t)row * DK + n2 * 16 + col] = (_Float16)(acc[n2][r] * inv[r]);
    }
  }
}

// ---------------------------------------------------------------------------
// Kernel 3: out = ctx @ Wo^T + bo, ctx f16 [B,H,S,DK] gathered, f32 out [B,S,D]
// Same 256-thread / 128x128-tile / 8-WMMA-per-interval shape as kernel 1.
// ---------------------------------------------------------------------------
__global__ __launch_bounds__(256) void out_proj_kernel(
    const _Float16* __restrict__ ctx, const float* __restrict__ W,
    const float* __restrict__ bias, float* __restrict__ out,
    int M, int N, int K, int S) {
  __shared__ __align__(16) _Float16 As[128 * LDT];
  __shared__ __align__(16) _Float16 Ws[128 * LDT];

  const int ntile = blockIdx.x * 128;
  const int mtile = blockIdx.y * 128;
  const int t = threadIdx.x;
  const int wave = t >> 5;
  const int lane = t & 31;
  const int mw = wave >> 1;
  const int nw = wave & 1;

  v8f acc[2][4];
#pragma unroll
  for (int g = 0; g < 2; ++g)
#pragma unroll
    for (int i = 0; i < 4; ++i) acc[g][i] = vzero();

  const int srow = t >> 1;
  const int scol = (t & 1) << 4;
  const int m0 = mtile + srow;
  const int b0 = m0 / S, s0 = m0 - b0 * S;

  for (int kb = 0; kb < K; kb += 32) {
    {
      // A from head-major ctx: k = kb+c -> head kb>>6, dim (kb&63)+c (contiguous)
      const _Float16* sa =
          ctx + (((size_t)b0 * HEADS + (kb >> 6)) * S + s0) * DK + (kb & 63) + scol;
      *(v8h*)(As + srow * LDT + scol) = *(const v8h*)sa;
      *(v8h*)(As + srow * LDT + scol + 8) = *(const v8h*)(sa + 8);

      const float* sw = W + (size_t)(ntile + srow) * K + kb + scol;
      _Float16* dw = Ws + srow * LDT + scol;
#pragma unroll
      for (int i = 0; i < 16; i += 4) {
        float4 fw = *(const float4*)(sw + i);
        dw[i + 0] = (_Float16)fw.x; dw[i + 1] = (_Float16)fw.y;
        dw[i + 2] = (_Float16)fw.z; dw[i + 3] = (_Float16)fw.w;
      }
      if (kb + 32 < K) __builtin_prefetch(sw + 32, 0, 0);
    }
    __syncthreads();

    v16h a0 = load_frag_a(As + (mw * 32) * LDT, LDT);
    v16h a1 = load_frag_a(As + (mw * 32 + 16) * LDT, LDT);
#pragma unroll
    for (int ns = 0; ns < 4; ++ns) {
      v16h b = load_frag_b(Ws + (nw * 64 + ns * 16) * LDT, LDT);
      acc[0][ns] = wmma16(a0, b, acc[0][ns]);
      acc[1][ns] = wmma16(a1, b, acc[1][ns]);
    }
    __syncthreads();
  }

  const int hi = lane >> 4;
  const int col = lane & 15;
#pragma unroll
  for (int g = 0; g < 2; ++g) {
    const int mrow = mtile + mw * 32 + g * 16;
#pragma unroll
    for (int ns = 0; ns < 4; ++ns) {
      const int n = ntile + nw * 64 + ns * 16 + col;
      const float bv = bias[n];
#pragma unroll
      for (int r = 0; r < 8; ++r) {
        const int m = mrow + (hi << 3) + r;
        out[(size_t)m * N + n] = acc[g][ns][r] + bv;
      }
    }
  }
}

// ---------------------------------------------------------------------------
// launcher
// ---------------------------------------------------------------------------
extern "C" void kernel_launch(void* const* d_in, const int* in_sizes, int n_in,
                              void* d_out, int out_size, void* d_ws, size_t ws_size,
                              hipStream_t stream) {
  const int S = 2048, D = 1024;
  const int B = in_sizes[0] / (S * D);
  const int M = B * S;

  const float* query = (const float*)d_in[0];
  const float* key   = (const float*)d_in[1];
  const float* value = (const float*)d_in[2];
  const float* Wq = (const float*)d_in[3]; const float* bq = (const float*)d_in[4];
  const float* Wk = (const float*)d_in[5]; const float* bk = (const float*)d_in[6];
  const float* Wv = (const float*)d_in[7]; const float* bv = (const float*)d_in[8];
  const float* Wo = (const float*)d_in[9]; const float* bo = (const float*)d_in[10];
  float* out = (float*)d_out;

  const size_t per = (size_t)M * D;  // elements per f16 tensor
  _Float16* q16 = (_Float16*)d_ws;
  _Float16* k16 = q16 + per;
  _Float16* v16 = k16 + per;
  _Float16* c16 = v16 + per;

  dim3 gP(D / 128, M / 128), bP(256);
  qkv_proj_kernel<<<gP, bP, 0, stream>>>(query, Wq, bq, q16, M, D, D, S);
  qkv_proj_kernel<<<gP, bP, 0, stream>>>(key,   Wk, bk, k16, M, D, D, S);
  qkv_proj_kernel<<<gP, bP, 0, stream>>>(value, Wv, bv, v16, M, D, D, S);

  dim3 gF(S / 128, B * HEADS), bF(256);
  flash_attn_kernel<<<gF, bF, 0, stream>>>(q16, k16, v16, c16, S);

  out_proj_kernel<<<gP, bP, 0, stream>>>(c16, Wo, bo, out, M, D, D, S);
}
